// MambaLayer_74852690034996
// MI455X (gfx1250) — matrix operand order repaired and checked by reading, not compiled, not run on hardware
//
#include <hip/hip_runtime.h>

// ---------------------------------------------------------------------------
// MI455X (gfx1250) implementation of the Mamba-3D block.
// All matrix-shaped work runs through v_wmma_f32_16x16x32_bf16 (bf16 operands,
// f32 accumulation). Activations live in "T layout": [channel][b*L + l],
// with NBL = 8192 columns, so GEMM B-operands are row-contiguous per k.
// GEMMs use 32x64 macro-tiles per wave (8 WMMAs per K-step, A reused 4x,
// B reused 2x); the 3x3x3 conv uses 32x32 macro-tiles (4 WMMAs per tap-step).
// ---------------------------------------------------------------------------

typedef __bf16 bf16_t;
typedef __attribute__((ext_vector_type(16))) __bf16 v16bf;
typedef __attribute__((ext_vector_type(8)))  __bf16 bf16x8;
typedef __attribute__((ext_vector_type(8)))  float  v8f;

static constexpr int S      = 16;
static constexpr int L      = 4096;           // S^3
static constexpr int BATCH  = 2;
static constexpr int NBL    = BATCH * L;      // 8192 columns
static constexpr int DIM    = 128;
static constexpr int DIN    = 256;            // d_inner
static constexpr int DSTATE = 16;
static constexpr int DTRANK = 8;
static constexpr int XPROJ_ROWS = DTRANK + 2 * DSTATE;  // 40
static constexpr int XPROJ_PAD  = 64;                    // padded to 32-multiple
static constexpr int MLPD   = 32;

#define DEV static __device__ __forceinline__

DEV v8f wmma_bf16(v16bf a, v16bf b, v8f c) {
  return __builtin_amdgcn_wmma_f32_16x16x32_bf16(
      /*neg_a=*/false, a, /*neg_b=*/false, b,
      /*c_mod=*/(short)0, c, /*reuse_a=*/false, /*reuse_b=*/false);
}

// A fragment: 16x32 bf16 tile, row-major source with leading dim K.
// lanes 0-15 : row = lane,    K = [k0..k0+7]  and [k0+16..k0+23]
// lanes 16-31: row = lane-16, K = [k0+8..k0+15] and [k0+24..k0+31]
DEV v16bf load_a_frag(const bf16_t* __restrict__ A, int K, int row, int k0, int lane) {
  const int half = lane >> 4;
  const bf16_t* p0 = A + (size_t)row * K + k0 + half * 8;
  bf16x8 lo = *(const bf16x8*)p0;
  bf16x8 hi = *(const bf16x8*)(p0 + 16);
  v16bf a;
#pragma unroll
  for (int i = 0; i < 8; ++i) { a[i] = lo[i]; a[8 + i] = hi[i]; }
  return a;
}

// B fragment: 32x16 bf16 tile from T-layout matrix [K][NBL].
// lane l holds column n = l % 16; K order matches load_a_frag.
DEV v16bf load_b_frag(const bf16_t* __restrict__ B, int k0, int colbase, int lane) {
  const bf16_t* p = B + (size_t)(k0 + (lane >> 4) * 8) * NBL + colbase + (lane & 15);
  v16bf b;
#pragma unroll
  for (int i = 0; i < 16; ++i) b[i] = p[(size_t)((i & 7) + (i >> 3) * 16) * NBL];
  return b;
}

// ---------------------------------------------------------------------------
// Generic WMMA GEMM: C[M][NBL] = A[M,K] * B[K][NBL] (+bias[row]).
// One wave computes a 32x64 macro-tile: 2 A-frags x 4 B-frags -> 8 WMMA/K-step.
// M must be a multiple of 32.
// ---------------------------------------------------------------------------
__global__ __launch_bounds__(128)
void k_gemm_T(const bf16_t* __restrict__ A, const bf16_t* __restrict__ B,
              const float* __restrict__ bias,
              float* __restrict__ Cf, bf16_t* __restrict__ Cb,
              int M, int K) {
  const int lane = threadIdx.x & 31;
  const int wid  = (blockIdx.x * blockDim.x + threadIdx.x) >> 5;
  const int ntilesN = NBL / 64;                 // 128 macro col tiles
  const int mt = wid / ntilesN;
  const int nt = wid % ntilesN;
  if (mt * 32 >= M) return;
  const int r0 = mt * 32;
  const int c0 = nt * 64;
  v8f acc[2][4] = {};
  for (int k0 = 0; k0 < K; k0 += 32) {
    if (k0 + 32 < K)
      __builtin_prefetch(B + (size_t)(k0 + 32 + lane) * NBL + c0, 0, 1);
    v16bf a0 = load_a_frag(A, K, r0 + (lane & 15), k0, lane);
    v16bf a1 = load_a_frag(A, K, r0 + 16 + (lane & 15), k0, lane);
    v16bf bf[4];
#pragma unroll
    for (int j = 0; j < 4; ++j) bf[j] = load_b_frag(B, k0, c0 + 16 * j, lane);
#pragma unroll
    for (int j = 0; j < 4; ++j) {
      acc[0][j] = wmma_bf16(a0, bf[j], acc[0][j]);
      acc[1][j] = wmma_bf16(a1, bf[j], acc[1][j]);
    }
  }
#pragma unroll
  for (int i = 0; i < 2; ++i) {
    const int mbase = r0 + i * 16 + (lane >> 4) * 8;
#pragma unroll
    for (int j = 0; j < 4; ++j) {
      const int n = c0 + j * 16 + (lane & 15);
#pragma unroll
      for (int r = 0; r < 8; ++r) {
        const int m = mbase + r;
        float v = acc[i][j][r] + (bias ? bias[m] : 0.0f);
        if (Cf) Cf[(size_t)m * NBL + n] = v;
        if (Cb) Cb[(size_t)m * NBL + n] = (bf16_t)v;
      }
    }
  }
}

// ---------------------------------------------------------------------------
// 3x3x3 conv as 27-tap WMMA GEMM. X is [128][NBL] bf16 in T layout; a 16-col
// tile is exactly one x-row (fixed b,z,y); a 32-aligned 32-col macro tile is
// two consecutive y-rows of the same z (y0 is even). dz/dy shifts select
// whole rows and dx=+-1 is a shifted column load with edge zeroing.
// Wp is tap-major packed weights: [27][O=128][I=128].
// One wave: 32 output channels x 32 columns -> 4 WMMAs per tap-K-step.
// ---------------------------------------------------------------------------
__global__ __launch_bounds__(128)
void k_conv3d_wmma(const bf16_t* __restrict__ Wp, const bf16_t* __restrict__ X,
                   const float* __restrict__ bias, float* __restrict__ Cf) {
  const int lane = threadIdx.x & 31;
  const int wid  = (blockIdx.x * blockDim.x + threadIdx.x) >> 5;
  const int ntilesN = NBL / 32;                 // 256 macro col tiles
  const int mt = wid / ntilesN;                 // 0..3 (32 rows each)
  const int nt = wid % ntilesN;
  const int colbase = nt * 32;
  const int b   = colbase >> 12;
  const int rem = colbase & (L - 1);
  const int z   = rem >> 8;
  const int y0  = (rem >> 4) & 15;              // even
  const int arow0 = mt * 32 + (lane & 15);
  v8f acc[2][2] = {};
  for (int t = 0; t < 27; ++t) {
    const int dz = t / 9 - 1;
    const int dy = (t / 3) % 3 - 1;
    const int dx = t % 3 - 1;
    const int zz = z + dz;
    if ((unsigned)zz >= 16u) continue;
    int rowbase[2];
    int vmask = 0;
#pragma unroll
    for (int j = 0; j < 2; ++j) {
      const int yy = y0 + j + dy;
      if ((unsigned)yy < 16u) {
        vmask |= (1 << j);
        rowbase[j] = b * L + (zz << 8) + (yy << 4);
      } else {
        rowbase[j] = 0;
      }
    }
    if (!vmask) continue;
    const bf16_t* Wt = Wp + (size_t)t * DIM * DIM;
    for (int k0 = 0; k0 < DIM; k0 += 32) {
      v16bf a0 = load_a_frag(Wt, DIM, arow0, k0, lane);
      v16bf a1 = load_a_frag(Wt, DIM, arow0 + 16, k0, lane);
#pragma unroll
      for (int j = 0; j < 2; ++j) {
        if (!(vmask & (1 << j))) continue;
        const int cn = (lane & 15) + dx;
        const bf16_t* bp = X + (size_t)(k0 + (lane >> 4) * 8) * NBL + rowbase[j] + cn;
        v16bf bv;
#pragma unroll
        for (int i = 0; i < 16; ++i)
          bv[i] = ((unsigned)cn < 16u) ? bp[(size_t)((i & 7) + (i >> 3) * 16) * NBL] : (bf16_t)0.0f;
        acc[0][j] = wmma_bf16(a0, bv, acc[0][j]);
        acc[1][j] = wmma_bf16(a1, bv, acc[1][j]);
      }
    }
  }
#pragma unroll
  for (int i = 0; i < 2; ++i) {
    const int mbase = mt * 32 + i * 16 + (lane >> 4) * 8;
#pragma unroll
    for (int j = 0; j < 2; ++j) {
      const int n = colbase + j * 16 + (lane & 15);
#pragma unroll
      for (int r = 0; r < 8; ++r) {
        const int m = mbase + r;
        Cf[(size_t)m * NBL + n] = acc[i][j][r] + bias[m];
      }
    }
  }
}

// ---------------------------------------------------------------------------
// Final pointwise GEMM: out = outT + c8 @ attn + c8_b, written in (B,C,L).
// 32x64 macro-tile per wave, M = 128, K = 128.
// ---------------------------------------------------------------------------
__global__ __launch_bounds__(128)
void k_gemm_c8_final(const bf16_t* __restrict__ A, const bf16_t* __restrict__ B,
                     const float* __restrict__ bias, const float* __restrict__ outT,
                     float* __restrict__ out) {
  const int lane = threadIdx.x & 31;
  const int wid  = (blockIdx.x * blockDim.x + threadIdx.x) >> 5;
  const int ntilesN = NBL / 64;
  const int mt = wid / ntilesN;                 // 0..3
  const int nt = wid % ntilesN;
  const int r0 = mt * 32;
  const int c0 = nt * 64;
  v8f acc[2][4] = {};
  for (int k0 = 0; k0 < DIM; k0 += 32) {
    v16bf a0 = load_a_frag(A, DIM, r0 + (lane & 15), k0, lane);
    v16bf a1 = load_a_frag(A, DIM, r0 + 16 + (lane & 15), k0, lane);
    v16bf bf[4];
#pragma unroll
    for (int j = 0; j < 4; ++j) bf[j] = load_b_frag(B, k0, c0 + 16 * j, lane);
#pragma unroll
    for (int j = 0; j < 4; ++j) {
      acc[0][j] = wmma_bf16(a0, bf[j], acc[0][j]);
      acc[1][j] = wmma_bf16(a1, bf[j], acc[1][j]);
    }
  }
#pragma unroll
  for (int i = 0; i < 2; ++i) {
    const int mbase = r0 + i * 16 + (lane >> 4) * 8;
#pragma unroll
    for (int j = 0; j < 4; ++j) {
      const int col = c0 + j * 16 + (lane & 15);
      const int bb  = col >> 12;
      const int l   = col & (L - 1);
#pragma unroll
      for (int r = 0; r < 8; ++r) {
        const int o = mbase + r;
        out[((size_t)bb * DIM + o) * L + l] =
            acc[i][j][r] + bias[o] + outT[(size_t)o * NBL + col];
      }
    }
  }
}

// ---------------------------------------------------------------------------
// Elementwise / reduction kernels
// ---------------------------------------------------------------------------
__global__ void k_embed(const float* __restrict__ x, const float* __restrict__ pos,
                        float* __restrict__ xflatT) {
  int idx = blockIdx.x * blockDim.x + threadIdx.x;
  if (idx >= DIM * NBL) return;
  const int c   = idx >> 13;            // NBL = 2^13
  const int col = idx & (NBL - 1);
  const int b   = col >> 12;
  const int l   = col & (L - 1);
  xflatT[idx] = x[((size_t)b * DIM + c) * L + l] + pos[(size_t)l * DIM + c];
}

// LayerNorm over 128 channels per column, T layout. block (32 cols, 8 partials)
__global__ __launch_bounds__(256)
void k_layernorm_T(const float* __restrict__ in, const float* __restrict__ w,
                   const float* __restrict__ bvec,
                   float* __restrict__ outf, bf16_t* __restrict__ outb) {
  __shared__ float s_sum[8][32];
  __shared__ float s_sq[8][32];
  __shared__ float s_mean[32], s_istd[32];
  const int tx = threadIdx.x, ty = threadIdx.y;
  const int col = blockIdx.x * 32 + tx;
  float vals[16];
  float sum = 0.0f, sq = 0.0f;
#pragma unroll
  for (int j = 0; j < 16; ++j) {
    const int c = ty + j * 8;
    const float v = in[(size_t)c * NBL + col];
    vals[j] = v; sum += v; sq += v * v;
  }
  s_sum[ty][tx] = sum; s_sq[ty][tx] = sq;
  __syncthreads();
  if (ty == 0) {
    float s = 0.0f, q = 0.0f;
#pragma unroll
    for (int j = 0; j < 8; ++j) { s += s_sum[j][tx]; q += s_sq[j][tx]; }
    const float m = s / 128.0f;
    s_mean[tx] = m;
    s_istd[tx] = rsqrtf(q / 128.0f - m * m + 1e-5f);
  }
  __syncthreads();
  const float m = s_mean[tx], istd = s_istd[tx];
#pragma unroll
  for (int j = 0; j < 16; ++j) {
    const int c = ty + j * 8;
    const float v = (vals[j] - m) * istd * w[c] + bvec[c];
    if (outf) outf[(size_t)c * NBL + col] = v;
    if (outb) outb[(size_t)c * NBL + col] = (bf16_t)v;
  }
}

// depthwise causal conv1d + SiLU; rev=1 handles the flipped-sequence branch
__global__ void k_conv1d_silu(const float* __restrict__ xz, const float* __restrict__ w,
                              const float* __restrict__ bias, int rev,
                              float* __restrict__ of, bf16_t* __restrict__ ob) {
  int idx = blockIdx.x * blockDim.x + threadIdx.x;
  if (idx >= DIN * NBL) return;
  const int d   = idx >> 13;
  const int col = idx & (NBL - 1);
  const int b   = col >> 12;
  const int l   = col & (L - 1);
  const float* row = xz + (size_t)d * NBL + (size_t)b * L;
  float acc = bias[d];
#pragma unroll
  for (int j = 0; j < 4; ++j) {
    const int lp = rev ? (l + 3 - j) : (l - 3 + j);
    const float v = ((unsigned)lp < (unsigned)L) ? row[lp] : 0.0f;
    acc += w[d * 4 + j] * v;
  }
  const float s = acc / (1.0f + __expf(-acc));
  of[idx] = s;
  ob[idx] = (bf16_t)s;
}

__global__ void k_dtproj(const float* __restrict__ xdbl, const float* __restrict__ w,
                         const float* __restrict__ bias, float* __restrict__ dt) {
  int idx = blockIdx.x * blockDim.x + threadIdx.x;
  if (idx >= DIN * NBL) return;
  const int d   = idx >> 13;
  const int col = idx & (NBL - 1);
  float acc = bias[d];
#pragma unroll
  for (int k = 0; k < DTRANK; ++k)
    acc += w[d * DTRANK + k] * xdbl[(size_t)k * NBL + col];
  dt[idx] = (acc > 20.0f) ? acc : log1pf(__expf(acc));
}

// selective scan: one (b,d) row per half-wave; lane = state index n.
// rev pass accumulates into y and emits bf16 for out_proj.
__global__ __launch_bounds__(128)
void k_scan(const float* __restrict__ dtf, const float* __restrict__ u,
            const float* __restrict__ xdbl, const float* __restrict__ xz,
            const float* __restrict__ A_log, const float* __restrict__ Dvec,
            int rev, float* __restrict__ y, bf16_t* __restrict__ yb) {
  const int lane = threadIdx.x & 31;
  const int wid  = (blockIdx.x * blockDim.x + threadIdx.x) >> 5;
  const int pair = wid * 2 + (lane >> 4);   // 0..511 = (b,d)
  const int b = pair >> 8;
  const int d = pair & 255;
  const int n = lane & 15;
  const float A  = -__expf(A_log[d * DSTATE + n]);
  const float Dd = Dvec[d];
  const float* dtrow = dtf  + (size_t)d * NBL + (size_t)b * L;
  const float* urow  = u    + (size_t)d * NBL + (size_t)b * L;
  const float* Brow  = xdbl + (size_t)(DTRANK + n) * NBL + (size_t)b * L;
  const float* Crow  = xdbl + (size_t)(DTRANK + DSTATE + n) * NBL + (size_t)b * L;
  const float* zrow  = xz   + (size_t)(DIN + d) * NBL + (size_t)b * L;
  float h = 0.0f;
  for (int t = 0; t < L; ++t) {
    const int p = rev ? (L - 1 - t) : t;
    const float delta = dtrow[p];
    const float uu    = urow[p];
    h = __expf(delta * A) * h + delta * Brow[p] * uu;
    float prod = h * Crow[p];
    prod += __shfl_xor(prod, 1, 16);
    prod += __shfl_xor(prod, 2, 16);
    prod += __shfl_xor(prod, 4, 16);
    prod += __shfl_xor(prod, 8, 16);
    if (n == 0) {
      float yv = prod + Dd * uu;
      const float zz = zrow[p];
      yv *= zz / (1.0f + __expf(-zz));
      const size_t oi = (size_t)d * NBL + (size_t)b * L + p;
      if (rev) {
        yv += y[oi];
        y[oi]  = yv;
        yb[oi] = (bf16_t)yv;
      } else {
        y[oi] = yv;
      }
    }
  }
}

__global__ void k_gelu_pack(const float* __restrict__ in, bf16_t* __restrict__ out, int n) {
  int idx = blockIdx.x * blockDim.x + threadIdx.x;
  if (idx >= n) return;
  const float x = in[idx];
  out[idx] = (bf16_t)(0.5f * x * (1.0f + erff(x * 0.70710678118f)));
}

__global__ void k_combine(const float* __restrict__ xflatT, const float* __restrict__ gamma,
                          const float* __restrict__ xm, const float* __restrict__ xs,
                          float* __restrict__ outT, bf16_t* __restrict__ outb) {
  int idx = blockIdx.x * blockDim.x + threadIdx.x;
  if (idx >= DIM * NBL) return;
  const int c = idx >> 13;
  const float v = xflatT[idx] + gamma[c] * (xm[idx] + xs[idx]);
  outT[idx] = v;
  outb[idx] = (bf16_t)v;
}

__global__ __launch_bounds__(256)
void k_bn_reduce(const float* __restrict__ in, float* __restrict__ mean,
                 float* __restrict__ istd) {
  __shared__ float ssum[256], ssq[256];
  const int c = blockIdx.x;
  const float* row = in + (size_t)c * NBL;
  float s = 0.0f, q = 0.0f;
  for (int i = threadIdx.x; i < NBL; i += 256) { const float v = row[i]; s += v; q += v * v; }
  ssum[threadIdx.x] = s; ssq[threadIdx.x] = q;
  __syncthreads();
  for (int st = 128; st > 0; st >>= 1) {
    if ((int)threadIdx.x < st) {
      ssum[threadIdx.x] += ssum[threadIdx.x + st];
      ssq[threadIdx.x]  += ssq[threadIdx.x + st];
    }
    __syncthreads();
  }
  if (threadIdx.x == 0) {
    const float m = ssum[0] / (float)NBL;
    mean[c] = m;
    istd[c] = rsqrtf(ssq[0] / (float)NBL - m * m + 1e-5f);
  }
}

__global__ void k_bn_lrelu_pack(const float* __restrict__ in, const float* __restrict__ mean,
                                const float* __restrict__ istd, const float* __restrict__ w,
                                const float* __restrict__ bvec, bf16_t* __restrict__ out) {
  int idx = blockIdx.x * blockDim.x + threadIdx.x;
  if (idx >= DIM * NBL) return;
  const int c = idx >> 13;
  float v = (in[idx] - mean[c]) * istd[c] * w[c] + bvec[c];
  v = (v >= 0.0f) ? v : 0.01f * v;
  out[idx] = (bf16_t)v;
}

__global__ void k_bn2_attn(const float* __restrict__ a2, const float* __restrict__ mean,
                           const float* __restrict__ istd, const float* __restrict__ w,
                           const float* __restrict__ bvec, const float* __restrict__ outT,
                           bf16_t* __restrict__ attn) {
  int idx = blockIdx.x * blockDim.x + threadIdx.x;
  if (idx >= DIM * NBL) return;
  const int c = idx >> 13;
  float v = (a2[idx] - mean[c]) * istd[c] * w[c] + bvec[c] + outT[idx];
  v = (v >= 0.0f) ? v : 0.01f * v;
  attn[idx] = (bf16_t)v;
}

// ---------------------------------------------------------------------------
// Weight packing
// ---------------------------------------------------------------------------
__global__ void k_pack_bf16(const float* __restrict__ src, bf16_t* __restrict__ dst, int n) {
  int idx = blockIdx.x * blockDim.x + threadIdx.x;
  if (idx < n) dst[idx] = (bf16_t)src[idx];
}

__global__ void k_pack_xproj(const float* __restrict__ src, bf16_t* __restrict__ dst) {
  int idx = blockIdx.x * blockDim.x + threadIdx.x;
  if (idx >= XPROJ_PAD * DIN) return;
  const int row = idx / DIN, col = idx % DIN;
  dst[idx] = (bf16_t)((row < XPROJ_ROWS) ? src[row * DIN + col] : 0.0f);
}

// (O,I,3,3,3) -> [tap=27][O][I]
__global__ void k_pack_conv(const float* __restrict__ src, bf16_t* __restrict__ dst) {
  int idx = blockIdx.x * blockDim.x + threadIdx.x;
  if (idx >= 27 * DIM * DIM) return;
  const int t = idx / (DIM * DIM);
  const int r = idx % (DIM * DIM);         // o*128 + i
  dst[idx] = (bf16_t)src[(size_t)r * 27 + t];
}

// ---------------------------------------------------------------------------
// Host orchestration
// ---------------------------------------------------------------------------
extern "C" void kernel_launch(void* const* d_in, const int* in_sizes, int n_in,
                              void* d_out, int out_size, void* d_ws, size_t ws_size,
                              hipStream_t stream) {
  const float* x          = (const float*)d_in[0];
  const float* pos        = (const float*)d_in[1];
  const float* ln_w       = (const float*)d_in[2];
  const float* ln_b       = (const float*)d_in[3];
  const float* ln1_w      = (const float*)d_in[4];
  const float* ln1_b      = (const float*)d_in[5];
  const float* ln2_w      = (const float*)d_in[6];
  const float* ln2_b      = (const float*)d_in[7];
  const float* gamma      = (const float*)d_in[8];
  const float* in_proj_w  = (const float*)d_in[9];
  const float* out_proj_w = (const float*)d_in[10];
  const float* conv1d_w   = (const float*)d_in[11];
  const float* conv1d_b   = (const float*)d_in[12];
  const float* x_proj_w   = (const float*)d_in[13];
  const float* dt_proj_w  = (const float*)d_in[14];
  const float* dt_proj_b  = (const float*)d_in[15];
  const float* A_log      = (const float*)d_in[16];
  const float* D_f        = (const float*)d_in[17];
  const float* conv1d_bw  = (const float*)d_in[18];
  const float* conv1d_bb  = (const float*)d_in[19];
  const float* x_proj_bw  = (const float*)d_in[20];
  const float* dt_proj_bw = (const float*)d_in[21];
  const float* dt_proj_bbias = (const float*)d_in[22];
  const float* A_b_log    = (const float*)d_in[23];
  const float* D_b        = (const float*)d_in[24];
  const float* fc1_w      = (const float*)d_in[25];
  const float* fc1_b      = (const float*)d_in[26];
  const float* fc2_w      = (const float*)d_in[27];
  const float* fc2_b      = (const float*)d_in[28];
  const float* c1_w       = (const float*)d_in[29];
  const float* c1_b       = (const float*)d_in[30];
  const float* bn1_w      = (const float*)d_in[31];
  const float* bn1_b      = (const float*)d_in[32];
  const float* c2_w       = (const float*)d_in[33];
  const float* c2_b       = (const float*)d_in[34];
  const float* bn2_w      = (const float*)d_in[35];
  const float* bn2_b      = (const float*)d_in[36];
  const float* c8_w       = (const float*)d_in[37];
  const float* c8_b       = (const float*)d_in[38];
  float* out = (float*)d_out;
  (void)in_sizes; (void)n_in; (void)out_size; (void)ws_size;

  char* ws = (char*)d_ws;
  size_t off = 0;
  auto alloc = [&](size_t bytes) -> char* {
    char* p = ws + off;
    off += (bytes + 255) & ~(size_t)255;
    return p;
  };
  const size_t CN = (size_t)DIM * NBL;     // 1,048,576

  // bf16 weight pool
  bf16_t* w_inproj  = (bf16_t*)alloc((size_t)2 * DIN * DIM * 2);
  bf16_t* w_outproj = (bf16_t*)alloc((size_t)DIM * DIN * 2);
  bf16_t* w_xpf     = (bf16_t*)alloc((size_t)XPROJ_PAD * DIN * 2);
  bf16_t* w_xpb     = (bf16_t*)alloc((size_t)XPROJ_PAD * DIN * 2);
  bf16_t* w_fc1     = (bf16_t*)alloc((size_t)MLPD * DIM * 2);
  bf16_t* w_fc2     = (bf16_t*)alloc((size_t)DIM * MLPD * 2);
  bf16_t* w_c8      = (bf16_t*)alloc((size_t)DIM * DIM * 2);
  bf16_t* w_cv1     = (bf16_t*)alloc((size_t)27 * DIM * DIM * 2);
  bf16_t* w_cv2     = (bf16_t*)alloc((size_t)27 * DIM * DIM * 2);
  // activations
  float*  xflatT = (float*)alloc(CN * 4);
  bf16_t* xnbf   = (bf16_t*)alloc(CN * 2);
  float*  xz     = (float*)alloc((size_t)2 * DIN * NBL * 4);
  float*  xconvf = (float*)alloc((size_t)DIN * NBL * 4);
  bf16_t* xconvb = (bf16_t*)alloc((size_t)DIN * NBL * 2);
  float*  xdbl   = (float*)alloc((size_t)XPROJ_PAD * NBL * 4);
  float*  dtf    = (float*)alloc((size_t)DIN * NBL * 4);
  float*  yacc   = (float*)alloc((size_t)DIN * NBL * 4);
  bf16_t* ybf    = (bf16_t*)alloc((size_t)DIN * NBL * 2);
  float*  mamba  = (float*)alloc(CN * 4);
  float*  xm     = (float*)alloc(CN * 4);
  float*  xs     = (float*)alloc(CN * 4);
  float*  h1     = (float*)alloc((size_t)MLPD * NBL * 4);
  bf16_t* h1bf   = (bf16_t*)alloc((size_t)MLPD * NBL * 2);
  bf16_t* h2bf   = (bf16_t*)alloc(CN * 2);
  float*  outT   = (float*)alloc(CN * 4);
  bf16_t* outbf  = (bf16_t*)alloc(CN * 2);
  float*  a1     = (float*)alloc(CN * 4);
  bf16_t* abf    = (bf16_t*)alloc(CN * 2);
  float*  a2     = (float*)alloc(CN * 4);
  bf16_t* attnbf = (bf16_t*)alloc(CN * 2);
  float*  stats  = (float*)alloc(512 * 4);
  float *mean1 = stats, *istd1 = stats + 128, *mean2 = stats + 256, *istd2 = stats + 384;

  auto ew = [](int n) { return dim3((unsigned)((n + 255) / 256)); };

  // ---- pack weights ----
  k_pack_bf16 <<<ew(2*DIN*DIM),   256, 0, stream>>>(in_proj_w,  w_inproj,  2*DIN*DIM);
  k_pack_bf16 <<<ew(DIM*DIN),     256, 0, stream>>>(out_proj_w, w_outproj, DIM*DIN);
  k_pack_xproj<<<ew(XPROJ_PAD*DIN), 256, 0, stream>>>(x_proj_w,  w_xpf);
  k_pack_xproj<<<ew(XPROJ_PAD*DIN), 256, 0, stream>>>(x_proj_bw, w_xpb);
  k_pack_bf16 <<<ew(MLPD*DIM),    256, 0, stream>>>(fc1_w, w_fc1, MLPD*DIM);
  k_pack_bf16 <<<ew(DIM*MLPD),    256, 0, stream>>>(fc2_w, w_fc2, DIM*MLPD);
  k_pack_bf16 <<<ew(DIM*DIM),     256, 0, stream>>>(c8_w,  w_c8,  DIM*DIM);
  k_pack_conv <<<ew(27*DIM*DIM),  256, 0, stream>>>(c1_w, w_cv1);
  k_pack_conv <<<ew(27*DIM*DIM),  256, 0, stream>>>(c2_w, w_cv2);

  // ---- embed + LN ----
  k_embed<<<ew((int)CN), 256, 0, stream>>>(x, pos, xflatT);
  k_layernorm_T<<<NBL/32, dim3(32, 8), 0, stream>>>(xflatT, ln_w, ln_b, nullptr, xnbf);

  auto gemm = [&](const bf16_t* A, const bf16_t* B, const float* bias,
                  float* Cf, bf16_t* Cb, int M, int K) {
    const int tiles = (M / 32) * (NBL / 64);   // 32x64 macro tiles, 4 waves/block
    k_gemm_T<<<tiles / 4, 128, 0, stream>>>(A, B, bias, Cf, Cb, M, K);
  };

  auto run_mamba = [&](const bf16_t* inbf, float* mout) {
    gemm(w_inproj, inbf, nullptr, xz, nullptr, 2 * DIN, DIM);
    // forward branch
    k_conv1d_silu<<<ew(DIN*NBL), 256, 0, stream>>>(xz, conv1d_w, conv1d_b, 0, xconvf, xconvb);
    gemm(w_xpf, xconvb, nullptr, xdbl, nullptr, XPROJ_PAD, DIN);
    k_dtproj<<<ew(DIN*NBL), 256, 0, stream>>>(xdbl, dt_proj_w, dt_proj_b, dtf);
    k_scan<<<64, 128, 0, stream>>>(dtf, xconvf, xdbl, xz, A_log, D_f, 0, yacc, ybf);
    // backward branch (flipped sequence handled by index reversal)
    k_conv1d_silu<<<ew(DIN*NBL), 256, 0, stream>>>(xz, conv1d_bw, conv1d_bb, 1, xconvf, xconvb);
    gemm(w_xpb, xconvb, nullptr, xdbl, nullptr, XPROJ_PAD, DIN);
    k_dtproj<<<ew(DIN*NBL), 256, 0, stream>>>(xdbl, dt_proj_bw, dt_proj_bbias, dtf);
    k_scan<<<64, 128, 0, stream>>>(dtf, xconvf, xdbl, xz, A_b_log, D_b, 1, yacc, ybf);
    gemm(w_outproj, ybf, nullptr, mout, nullptr, DIM, DIN);
  };

  // ---- mamba #1 on x_norm ----
  run_mamba(xnbf, mamba);
  k_layernorm_T<<<NBL/32, dim3(32, 8), 0, stream>>>(mamba, ln1_w, ln1_b, xm, nullptr);

  // ---- MLP (pointwise conv3d pair) ----
  gemm(w_fc1, xnbf, fc1_b, h1, nullptr, MLPD, DIM);
  k_gelu_pack<<<ew(MLPD*NBL), 256, 0, stream>>>(h1, h1bf, MLPD * NBL);
  gemm(w_fc2, h1bf, fc2_b, nullptr, h2bf, DIM, MLPD);

  // ---- mamba #2 on h ----
  run_mamba(h2bf, mamba);
  k_layernorm_T<<<NBL/32, dim3(32, 8), 0, stream>>>(mamba, ln2_w, ln2_b, xs, nullptr);

  // ---- residual combine ----
  k_combine<<<ew((int)CN), 256, 0, stream>>>(xflatT, gamma, xm, xs, outT, outbf);

  // ---- conv / bn / attention tail ----
  {
    const int ctiles = (DIM / 32) * (NBL / 32);   // 4 * 256 = 1024 waves
    k_conv3d_wmma<<<ctiles / 4, 128, 0, stream>>>(w_cv1, outbf, c1_b, a1);
    k_bn_reduce<<<DIM, 256, 0, stream>>>(a1, mean1, istd1);
    k_bn_lrelu_pack<<<ew((int)CN), 256, 0, stream>>>(a1, mean1, istd1, bn1_w, bn1_b, abf);
    k_conv3d_wmma<<<ctiles / 4, 128, 0, stream>>>(w_cv2, abf, c2_b, a2);
    k_bn_reduce<<<DIM, 256, 0, stream>>>(a2, mean2, istd2);
    k_bn2_attn<<<ew((int)CN), 256, 0, stream>>>(a2, mean2, istd2, bn2_w, bn2_b, outT, attnbf);
  }

  // ---- final pointwise GEMM + residual, write (B,C,D,H,W) ----
  {
    const int ftiles = (DIM / 32) * (NBL / 64);   // 4 * 128 = 512 waves
    k_gemm_c8_final<<<ftiles / 4, 128, 0, stream>>>(w_c8, attnbf, c8_b, outT, out);
  }
}
